// ConversationGAT_26637387170409
// MI455X (gfx1250) — matrix operand order repaired and checked by reading, not compiled, not run on hardware
//
#include <hip/hip_runtime.h>
#include <hip/hip_bf16.h>
#include <math.h>

typedef _Float16 f16;
typedef __attribute__((ext_vector_type(16))) _Float16 v16h;
typedef __attribute__((ext_vector_type(8)))  _Float16 v8h;
typedef __attribute__((ext_vector_type(8)))  float    v8f;

#define CDIV(a,b) (((a)+(b)-1)/(b))
#define HL 128      // LSTM hidden
#define G4 512      // 4*HL

// ---------------------------------------------------------------- utilities
__global__ void k_f32_to_f16(f16* __restrict__ dst, const float* __restrict__ src, int n){
  int i = blockIdx.x*blockDim.x + threadIdx.x;
  if (i < n) dst[i] = (f16)src[i];
}
__global__ void k_fill(float* __restrict__ p, float v, int n){
  int i = blockIdx.x*blockDim.x + threadIdx.x;
  if (i < n) p[i] = v;
}
__global__ void k_add_vec(float* __restrict__ c, const float* __restrict__ a,
                          const float* __restrict__ b, int n){
  int i = blockIdx.x*blockDim.x + threadIdx.x;
  if (i < n) c[i] = a[i] + b[i];
}
__global__ void k_init_bias(float* __restrict__ out, const float* __restrict__ bias, int n, int C){
  int i = blockIdx.x*blockDim.x + threadIdx.x;
  if (i < n) out[i] = bias[i % C];
}

// ---------------------------------------------------------------- WMMA GEMM
// C[M,N] = A[M,K](f16,row-major) * B[Npad,K]^T(f16,row-major "weights") + bias[N]
// Caller guarantees: M % 16 == 0, K % 64 == 0, B padded to align16(N) rows,
// 16B alignment everywhere, and CDIV(N,16) % NT == 0 (NT is compile-time).
// Each wave computes a 16x(16*NT) C panel: NT accumulators share one A
// fragment. All fragment loads for a K-step are issued before the first WMMA
// so the scheduler can use partial loadcnt waits (loads overlap matrix ops).
// No divergent control flow: EXEC is all-ones through every WMMA (ISA req).
template<int NT>
__global__ __launch_bounds__(256)
void k_wmma_gemm(const f16* __restrict__ A, const f16* __restrict__ B,
                 const float* __restrict__ bias, float* __restrict__ C,
                 int M, int N, int K)
{
  const int wave    = threadIdx.x >> 5;
  const int lane    = threadIdx.x & 31;
  const int tilesN  = (N + 15) >> 4;
  const int groupsN = tilesN / NT;         // exact by contract
  const int id  = blockIdx.x * 8 + wave;
  const int tM  = id / groupsN;
  const int tN0 = (id - tM * groupsN) * NT;
  if (tM * 16 >= M) return;                // whole-wave early exit only

  const int halfSel = lane >> 4;           // 0: lanes 0-15, 1: lanes 16-31
  const int aRow  = tM * 16 + (lane & 15);
  const int bRow0 = tN0 * 16 + (lane & 15);

  // A 16x32 f16 fragment: elems[0..7] = K k0..k0+7 (+8 for upper lanes),
  //                       elems[8..15] = K k0+16..k0+23 (+8 for upper lanes)
  const f16* ap  = A + (size_t)aRow  * K + halfSel * 8;
  // B 32x16 fragment: col = bRow, elems[0..7] = K k0..k0+7, elems[8..15] = K k0+16..k0+23 (+8 upper lanes)
  const f16* bp0 = B + (size_t)bRow0 * K + halfSel * 8;

  v8f acc[NT];
  #pragma unroll
  for (int nt = 0; nt < NT; ++nt) acc[nt] = (v8f){};

  #pragma unroll 2
  for (int k0 = 0; k0 < K; k0 += 32) {
    // ---- issue every load for this K-step up front ----
    v8h a0 = *(const v8h*)(ap + k0);
    v8h a1 = *(const v8h*)(ap + k0 + 16);
    v8h b0[NT], b1[NT];
    #pragma unroll
    for (int nt = 0; nt < NT; ++nt) {
      const f16* bp = bp0 + (size_t)nt * 16 * K + k0;
      b0[nt] = *(const v8h*)(bp);
      b1[nt] = *(const v8h*)(bp + 16);
    }
    // ---- consume: NT back-to-back WMMAs sharing one A fragment ----
    v16h a = __builtin_shufflevector(a0, a1, 0,1,2,3,4,5,6,7,8,9,10,11,12,13,14,15);
    #pragma unroll
    for (int nt = 0; nt < NT; ++nt) {
      v16h b = __builtin_shufflevector(b0[nt], b1[nt],
                                       0,1,2,3,4,5,6,7,8,9,10,11,12,13,14,15);
      acc[nt] = __builtin_amdgcn_wmma_f32_16x16x32_f16(false, a, false, b,
                                                       (short)0, acc[nt], false, false);
    }
  }

  // C/D layout: lanes 0-15: col=lane, VGPR v -> row v; lanes 16-31: row v+8
  const int rBase = tM * 16 + halfSel * 8;
  #pragma unroll
  for (int nt = 0; nt < NT; ++nt) {
    int col = (tN0 + nt) * 16 + (lane & 15);
    if (col < N) {
      float bv = bias ? bias[col] : 0.f;
      #pragma unroll
      for (int v = 0; v < 8; ++v)
        C[(size_t)(rBase + v) * N + col] = acc[nt][v] + bv;
    }
  }
}

// ---------------------------------------------------------------- LSTM recurrence
__device__ __forceinline__ float sigm(float x){ return 1.f/(1.f + __expf(-x)); }

// Persistent recurrence: blockIdx.x = 0 (forward) / 1 (backward).
// gates_* = precomputed x@Wih^T + bih + bhh, [T, 512]. Whh (f16) staged in LDS,
// transposed to [k][j] so lanes (consecutive j) hit consecutive banks.
__global__ __launch_bounds__(512)
void k_lstm_layer(const float* __restrict__ gates_f, const float* __restrict__ gates_b,
                  const f16* __restrict__ whh_f, const f16* __restrict__ whh_b,
                  float* __restrict__ hout, int T)
{
  extern __shared__ float smem[];
  float* zbuf = smem;              // 512
  float* hb   = smem + 512;        // 128
  float* cb   = smem + 640;        // 128
  f16*   wsh  = (f16*)(smem + 768);// [128][512] transposed Whh, 131072 B

  const int rev = blockIdx.x;
  const float* gates = rev ? gates_b : gates_f;
  const f16*   whh   = rev ? whh_b  : whh_f;
  const int outOff   = rev ? HL : 0;
  const int j = threadIdx.x;       // gate element 0..511

  for (int i = j; i < G4*HL; i += blockDim.x) {
    int r = i >> 7, k = i & (HL-1);      // source is row-major [512,128]
    wsh[k*G4 + r] = whh[i];
  }
  if (j < HL) { hb[j] = 0.f; cb[j] = 0.f; }
  __syncthreads();

  for (int s = 0; s < T; ++s) {
    const int t = rev ? (T-1-s) : s;
    float z = gates[(size_t)t*G4 + j];
    #pragma unroll 8
    for (int k = 0; k < HL; ++k) z += (float)wsh[k*G4 + j] * hb[k];
    zbuf[j] = z;
    __syncthreads();
    if (j < HL) {
      float zi = zbuf[j], zf = zbuf[HL + j], zg = zbuf[2*HL + j], zo = zbuf[3*HL + j];
      float c = sigm(zf)*cb[j] + sigm(zi)*tanhf(zg);
      float h = sigm(zo)*tanhf(c);
      cb[j] = c; hb[j] = h;
      hout[(size_t)t*(2*HL) + outOff + j] = h;
    }
    __syncthreads();
  }
}

// ---------------------------------------------------------------- GAT pieces
// Per-node head scores: ssrc[n,h] = sum_c h1[n,h,c]*a_src[h,c] (and a_dst).
__global__ void k_scores(const float* __restrict__ h, const float* __restrict__ a_src,
                         const float* __restrict__ a_dst, float* __restrict__ ssrc,
                         float* __restrict__ sdst, int N, int H, int C){
  int i = blockIdx.x*blockDim.x + threadIdx.x;
  if (i >= N*H) return;
  int hd = i % H, n = i / H;
  const float* hp = h + (size_t)n*H*C + hd*C;
  const float* as = a_src + hd*C;
  const float* ad = a_dst + hd*C;
  float s0 = 0.f, s1 = 0.f;
  for (int c = 0; c < C; ++c){ float v = hp[c]; s0 += v*as[c]; s1 += v*ad[c]; }
  ssrc[i] = s0; sdst[i] = s1;
}

// Fold a_e into We: V[h,d] = sum_c a_e[h,c] * We[h*C+c, d].  V padded to 16 rows (f16).
__global__ void k_buildV(f16* __restrict__ V, const float* __restrict__ ae,
                         const float* __restrict__ We, int H, int C, int D){
  int i = blockIdx.x*blockDim.x + threadIdx.x;
  if (i >= 16*D) return;
  int hd = i / D, d = i - hd*D;
  float s = 0.f;
  if (hd < H)
    for (int c = 0; c < C; ++c) s += ae[hd*C + c] * We[(size_t)(hd*C + c)*D + d];
  V[i] = (f16)s;
}

__device__ __forceinline__ void atomicMaxF(float* addr, float v){
  if (v >= 0.f) atomicMax((int*)addr, __float_as_int(v));
  else          atomicMin((unsigned int*)addr, (unsigned int)__float_as_int(v));
}

__global__ void k_alpha_max(const int* __restrict__ ei, const float* __restrict__ ssrc,
                            const float* __restrict__ sdst, const float* __restrict__ aen,
                            float* __restrict__ alpha, float* __restrict__ amax, int E){
  int i = blockIdx.x*blockDim.x + threadIdx.x;
  if (i >= E*8) return;
  int e = i >> 3, hd = i & 7;
  int s = ei[e], d = ei[E + e];
  float a = ssrc[s*8+hd] + sdst[d*8+hd] + aen[s*8+hd] + aen[d*8+hd];
  a = (a > 0.f) ? a : 0.2f*a;                    // leaky_relu(0.2)
  alpha[i] = a;
  atomicMaxF(&amax[d*8 + hd], a);
}

__global__ void k_exp_sum(const int* __restrict__ ei, float* __restrict__ alpha,
                          const float* __restrict__ amax, float* __restrict__ denom, int E){
  int i = blockIdx.x*blockDim.x + threadIdx.x;
  if (i >= E*8) return;
  int e = i >> 3, hd = i & 7;
  int d = ei[E + e];
  float ex = __expf(alpha[i] - amax[d*8 + hd]);
  alpha[i] = ex;                                  // overwrite in place
  atomicAdd(&denom[d*8 + hd], ex);
}

__global__ void k_msg(const int* __restrict__ ei, const float* __restrict__ alpha,
                      const float* __restrict__ denom, const float* __restrict__ h,
                      float* __restrict__ out, int E, int C){
  int i = blockIdx.x*blockDim.x + threadIdx.x;
  if (i >= E*8) return;
  int e = i >> 3, hd = i & 7;
  int s = ei[e], d = ei[E + e];
  float w = alpha[i] / (denom[d*8 + hd] + 1e-16f);
  const float* hp = h  + (size_t)s*(8*C) + hd*C;
  float*       op = out + (size_t)d*(8*C) + hd*C;
  for (int c = 0; c < C; ++c) atomicAdd(&op[c], w * hp[c]);
}

// ---------------------------------------------------------------- BatchNorm
__global__ void k_bn_stats(const float* __restrict__ x, float* __restrict__ stats,
                           int N, int C){
  int ch = threadIdx.x;                 // blockDim.x == C
  int rows = CDIV(N, (int)gridDim.x);
  int r0 = blockIdx.x * rows;
  int r1 = r0 + rows; if (r1 > N) r1 = N;
  float s = 0.f, q = 0.f;
  for (int r = r0; r < r1; ++r){ float v = x[(size_t)r*C + ch]; s += v; q += v*v; }
  atomicAdd(&stats[ch], s);
  atomicAdd(&stats[C + ch], q);
}

__global__ void k_bn_relu_f16(f16* __restrict__ y, const float* __restrict__ x,
                              const float* __restrict__ stats, const float* __restrict__ gamma,
                              const float* __restrict__ beta, int N, int C){
  int i = blockIdx.x*blockDim.x + threadIdx.x;
  if (i >= N*C) return;
  int ch = i % C;
  float invN = 1.f / (float)N;
  float m = stats[ch] * invN;
  float v = stats[C + ch] * invN - m*m;
  float t = gamma[ch] * (x[i] - m) * rsqrtf(v + 1e-5f) + beta[ch];
  y[i] = (f16)fmaxf(t, 0.f);
}

// ---------------------------------------------------------------- edge head
__global__ void k_head(float* __restrict__ out, const f16* __restrict__ h,
                       const int* __restrict__ ei, const float* __restrict__ ow,
                       const float* __restrict__ ob, int E){
  int e = blockIdx.x*blockDim.x + threadIdx.x;
  if (e >= E) return;
  int s = ei[e], d = ei[E + e];
  const f16* ps = h + (size_t)s*64;
  const f16* pd = h + (size_t)d*64;
  float a0 = ob[0], a1 = ob[1];
  #pragma unroll 8
  for (int c = 0; c < 64; ++c){
    float vs = (float)ps[c], vd = (float)pd[c];
    a0 += vs*ow[c]       + vd*ow[64 + c];
    a1 += vs*ow[128 + c] + vd*ow[192 + c];
  }
  out[(size_t)e*2]     = a0;
  out[(size_t)e*2 + 1] = a1;
}

// ---------------------------------------------------------------- host driver
extern "C" void kernel_launch(void* const* d_in, const int* in_sizes, int n_in,
                              void* d_out, int out_size, void* d_ws, size_t ws_size,
                              hipStream_t stream)
{
  const int N = 20000, E = 400000, DIN = 768, DEA = 256;

  const float* x     = (const float*)d_in[0];
  const int*   ei    = (const int*)  d_in[1];
  const float* eattr = (const float*)d_in[2];
  // LSTM params: 0=l0 fwd, 1=l0 bwd, 2=l1 fwd, 3=l1 bwd
  const float* Wih[4] = {(const float*)d_in[3], (const float*)d_in[7],
                         (const float*)d_in[11], (const float*)d_in[15]};
  const float* Whh[4] = {(const float*)d_in[4], (const float*)d_in[8],
                         (const float*)d_in[12], (const float*)d_in[16]};
  const float* bih[4] = {(const float*)d_in[5], (const float*)d_in[9],
                         (const float*)d_in[13], (const float*)d_in[17]};
  const float* bhh[4] = {(const float*)d_in[6], (const float*)d_in[10],
                         (const float*)d_in[14], (const float*)d_in[18]};
  const float* g1W   = (const float*)d_in[19]; const float* g1We  = (const float*)d_in[20];
  const float* g1as  = (const float*)d_in[21]; const float* g1ad  = (const float*)d_in[22];
  const float* g1ae  = (const float*)d_in[23]; const float* g1b   = (const float*)d_in[24];
  const float* g2W   = (const float*)d_in[25]; const float* g2We  = (const float*)d_in[26];
  const float* g2as  = (const float*)d_in[27]; const float* g2ad  = (const float*)d_in[28];
  const float* g2ae  = (const float*)d_in[29]; const float* g2b   = (const float*)d_in[30];
  const float* bn1g  = (const float*)d_in[31]; const float* bn1b  = (const float*)d_in[32];
  const float* bn2g  = (const float*)d_in[33]; const float* bn2b  = (const float*)d_in[34];
  const float* outW  = (const float*)d_in[35]; const float* outB  = (const float*)d_in[36];
  float* out = (float*)d_out;

  // bump allocator over workspace (256B aligned blocks)
  char* wbase = (char*)d_ws;
  size_t off = 0;
  auto alloc = [&](size_t bytes) -> void* {
    void* p = wbase + off;
    off = (off + bytes + 255) & ~(size_t)255;
    return p;
  };

  f16* xf16       = (f16*)alloc((size_t)N*DIN*2);
  f16* wihF[4]; wihF[0] = (f16*)alloc((size_t)G4*DIN*2); wihF[1] = (f16*)alloc((size_t)G4*DIN*2);
                wihF[2] = (f16*)alloc((size_t)G4*256*2); wihF[3] = (f16*)alloc((size_t)G4*256*2);
  f16* whhF[4]; for (int i = 0; i < 4; ++i) whhF[i] = (f16*)alloc((size_t)G4*HL*2);
  f16* w1f16      = (f16*)alloc((size_t)128*256*2);
  f16* w2f16      = (f16*)alloc((size_t)64*128*2);
  float* bsum[4]; for (int i = 0; i < 4; ++i) bsum[i] = (float*)alloc((size_t)G4*4);
  float* gF       = (float*)alloc((size_t)N*G4*4);    // reused for layer 0 and 1
  float* gB       = (float*)alloc((size_t)N*G4*4);
  float* hcat0    = (float*)alloc((size_t)N*256*4);
  f16*   hcat0h   = (f16*)  alloc((size_t)N*256*2);
  float* hcat1    = (float*)alloc((size_t)N*256*4);
  f16*   hcat1h   = (f16*)  alloc((size_t)N*256*2);
  f16*   eaf16    = (f16*)  alloc((size_t)N*DEA*2);
  float* h1       = (float*)alloc((size_t)N*128*4);
  float* ssrc     = (float*)alloc((size_t)N*8*4);     // reused for GAT2
  float* sdst     = (float*)alloc((size_t)N*8*4);
  float* aen      = (float*)alloc((size_t)N*8*4);
  f16*   Vf16     = (f16*)  alloc((size_t)16*256*2);  // zero rows >= H via k_buildV
  float* alpha    = (float*)alloc((size_t)E*8*4);     // reused for GAT2
  float* amax     = (float*)alloc((size_t)N*8*4);
  float* denom    = (float*)alloc((size_t)N*8*4);
  float* out1     = (float*)alloc((size_t)N*128*4);
  f16*   h1rh     = (f16*)  alloc((size_t)N*128*2);
  float* h2       = (float*)alloc((size_t)N*64*4);
  float* out2     = (float*)alloc((size_t)N*64*4);
  f16*   h2rh     = (f16*)  alloc((size_t)N*64*2);
  float* stats    = (float*)alloc((size_t)256*4);
  (void)ws_size; (void)in_sizes; (void)n_in; (void)out_size;

  auto conv = [&](f16* d, const float* s, int n){
    k_f32_to_f16<<<CDIV(n,256),256,0,stream>>>(d, s, n);
  };
  auto gemm = [&](const f16* A, const f16* B, const float* bias, float* C,
                  int M, int Nn, int K){
    int tilesN = CDIV(Nn, 16);
    if ((tilesN & 3) == 0) {
      int tiles = CDIV(M,16) * (tilesN / 4);
      k_wmma_gemm<4><<<CDIV(tiles,8),256,0,stream>>>(A, B, bias, C, M, Nn, K);
    } else {
      int tiles = CDIV(M,16) * tilesN;
      k_wmma_gemm<1><<<CDIV(tiles,8),256,0,stream>>>(A, B, bias, C, M, Nn, K);
    }
  };

  // --- precision conversions + fused biases ---
  conv(xf16, x, N*DIN);
  conv(wihF[0], Wih[0], G4*DIN); conv(wihF[1], Wih[1], G4*DIN);
  conv(wihF[2], Wih[2], G4*256); conv(wihF[3], Wih[3], G4*256);
  for (int i = 0; i < 4; ++i) {
    conv(whhF[i], Whh[i], G4*HL);
    k_add_vec<<<CDIV(G4,256),256,0,stream>>>(bsum[i], bih[i], bhh[i], G4);
  }
  conv(eaf16, eattr, N*DEA);
  conv(w1f16, g1W, 128*256);
  conv(w2f16, g2W, 64*128);

  const size_t lstmSmem = (512 + 128 + 128)*4 + (size_t)G4*HL*2;  // 134144 B < 320KB/WGP

  // --- LSTM layer 0: batched input GEMMs (WMMA) + persistent recurrence ---
  gemm(xf16, wihF[0], bsum[0], gF, N, G4, DIN);
  gemm(xf16, wihF[1], bsum[1], gB, N, G4, DIN);
  k_lstm_layer<<<2,512,lstmSmem,stream>>>(gF, gB, whhF[0], whhF[1], hcat0, N);
  conv(hcat0h, hcat0, N*256);

  // --- LSTM layer 1 ---
  gemm(hcat0h, wihF[2], bsum[2], gF, N, G4, 256);
  gemm(hcat0h, wihF[3], bsum[3], gB, N, G4, 256);
  k_lstm_layer<<<2,512,lstmSmem,stream>>>(gF, gB, whhF[2], whhF[3], hcat1, N);
  conv(hcat1h, hcat1, N*256);

  const int EB = CDIV(E*8, 256);

  // --- GAT layer 1 (heads=8, ch=16) ---
  gemm(hcat1h, w1f16, nullptr, h1, N, 128, 256);
  k_scores<<<CDIV(N*8,256),256,0,stream>>>(h1, g1as, g1ad, ssrc, sdst, N, 8, 16);
  k_buildV<<<CDIV(16*256,256),256,0,stream>>>(Vf16, g1ae, g1We, 8, 16, 256);
  gemm(eaf16, Vf16, nullptr, aen, N, 8, 256);    // node-level folded edge-attr scores
  k_fill<<<CDIV(N*8,256),256,0,stream>>>(amax, -INFINITY, N*8);
  k_fill<<<CDIV(N*8,256),256,0,stream>>>(denom, 0.f, N*8);
  k_init_bias<<<CDIV(N*128,256),256,0,stream>>>(out1, g1b, N*128, 128);
  k_alpha_max<<<EB,256,0,stream>>>(ei, ssrc, sdst, aen, alpha, amax, E);
  k_exp_sum  <<<EB,256,0,stream>>>(ei, alpha, amax, denom, E);
  k_msg      <<<EB,256,0,stream>>>(ei, alpha, denom, h1, out1, E, 16);
  k_fill<<<1,256,0,stream>>>(stats, 0.f, 256);
  k_bn_stats<<<64,128,0,stream>>>(out1, stats, N, 128);
  k_bn_relu_f16<<<CDIV(N*128,256),256,0,stream>>>(h1rh, out1, stats, bn1g, bn1b, N, 128);

  // --- GAT layer 2 (heads=8, ch=8) ---
  gemm(h1rh, w2f16, nullptr, h2, N, 64, 128);
  k_scores<<<CDIV(N*8,256),256,0,stream>>>(h2, g2as, g2ad, ssrc, sdst, N, 8, 8);
  k_buildV<<<CDIV(16*256,256),256,0,stream>>>(Vf16, g2ae, g2We, 8, 8, 256);
  gemm(eaf16, Vf16, nullptr, aen, N, 8, 256);
  k_fill<<<CDIV(N*8,256),256,0,stream>>>(amax, -INFINITY, N*8);
  k_fill<<<CDIV(N*8,256),256,0,stream>>>(denom, 0.f, N*8);
  k_init_bias<<<CDIV(N*64,256),256,0,stream>>>(out2, g2b, N*64, 64);
  k_alpha_max<<<EB,256,0,stream>>>(ei, ssrc, sdst, aen, alpha, amax, E);
  k_exp_sum  <<<EB,256,0,stream>>>(ei, alpha, amax, denom, E);
  k_msg      <<<EB,256,0,stream>>>(ei, alpha, denom, h2, out2, E, 8);
  k_fill<<<1,256,0,stream>>>(stats, 0.f, 256);
  k_bn_stats<<<64,64,0,stream>>>(out2, stats, N, 64);
  k_bn_relu_f16<<<CDIV(N*64,256),256,0,stream>>>(h2rh, out2, stats, bn2g, bn2b, N, 64);

  // --- per-edge classification head ---
  k_head<<<CDIV(E,256),256,0,stream>>>(out, h2rh, ei, outW, outB, E);
}
